// VRGCore_64476049047930
// MI455X (gfx1250) — matrix operand order repaired and checked by reading, not compiled, not run on hardware
//
#include <hip/hip_runtime.h>
#include <math.h>

typedef __attribute__((ext_vector_type(16))) _Float16 v16h;
typedef __attribute__((ext_vector_type(8)))  _Float16 v8h;
typedef __attribute__((ext_vector_type(8)))  float    v8f;

#define B_  128
#define R_  1024
#define A_  512
#define S_  196
#define SR_ 64

__device__ __forceinline__ float sigm(float x) { return 1.0f / (1.0f + expf(-x)); }

// ---------------------------------------------------------------------------
// Generic WMMA GEMM: C(MxN) = A(MxK) @ W(N,K)^T [+bias[n]] [+C]
// A, W are fp32 in HBM; converted to f16 while staging into LDS.
// Block tile 64x128, K-step 32, 256 threads = 8 waves (2x4), wave tile 32x32.
// ADA epilogue: instead of storing C, computes
//   t = tanh(acc + bias[n] + ho_embed[b*R + n]);  e[b*(S+1)+1+s] += t*ada_w[n]
// with row = b*S + s (fused adaptive-attention logits, cfe never materialized).
// ---------------------------------------------------------------------------
template <bool ADA>
__global__ __launch_bounds__(256) void gemm_kernel(
    const float* __restrict__ Ag, int lda,
    const float* __restrict__ Wg, int ldw,
    float* __restrict__ Cg, int ldc,
    const float* __restrict__ bias, int accum,
    int M, int N, int K,
    const float* __restrict__ ho_embed,
    const float* __restrict__ ada_w,
    float* __restrict__ e_ada)
{
  __shared__ _Float16 sA[64][40];
  __shared__ _Float16 sB[128][40];

  const int tid  = threadIdx.x;
  const int wave = tid >> 5;
  const int lane = tid & 31;
  const int wm   = wave >> 2;      // 0..1
  const int wn   = wave & 3;       // 0..3
  const int lh   = lane >> 4;      // lane half
  const int rc   = lane & 15;      // row (A) / col (B,C)
  const int bm0  = blockIdx.y * 64;
  const int bn0  = blockIdx.x * 128;

  v8f acc[2][2];
  for (int i = 0; i < 2; ++i)
    for (int j = 0; j < 2; ++j)
      for (int q = 0; q < 8; ++q) acc[i][j][q] = 0.0f;

  const int arow = tid >> 2;            // 0..63
  const int acol = (tid & 3) * 8;       // 0,8,16,24
  const int brow = tid >> 1;            // 0..127
  const int bcol = (tid & 1) * 16;      // 0,16
  const float* aptr = Ag + (size_t)(bm0 + arow) * lda + acol;
  const float* bptr = Wg + (size_t)(bn0 + brow) * ldw + bcol;

  for (int k0 = 0; k0 < K; k0 += 32) {
    // stage A tile (64x32) fp32 -> f16
    {
      const float4 a0 = *(const float4*)(aptr + k0);
      const float4 a1 = *(const float4*)(aptr + k0 + 4);
      v8h av = { (_Float16)a0.x, (_Float16)a0.y, (_Float16)a0.z, (_Float16)a0.w,
                 (_Float16)a1.x, (_Float16)a1.y, (_Float16)a1.z, (_Float16)a1.w };
      *(v8h*)&sA[arow][acol] = av;
    }
    // stage B tile (128x32) fp32 -> f16 (W rows are output-channel major)
    {
      const float4 b0 = *(const float4*)(bptr + k0);
      const float4 b1 = *(const float4*)(bptr + k0 + 4);
      const float4 b2 = *(const float4*)(bptr + k0 + 8);
      const float4 b3 = *(const float4*)(bptr + k0 + 12);
      v8h x0 = { (_Float16)b0.x, (_Float16)b0.y, (_Float16)b0.z, (_Float16)b0.w,
                 (_Float16)b1.x, (_Float16)b1.y, (_Float16)b1.z, (_Float16)b1.w };
      v8h x1 = { (_Float16)b2.x, (_Float16)b2.y, (_Float16)b2.z, (_Float16)b2.w,
                 (_Float16)b3.x, (_Float16)b3.y, (_Float16)b3.z, (_Float16)b3.w };
      *(v8h*)&sB[brow][bcol]     = x0;
      *(v8h*)&sB[brow][bcol + 8] = x1;
    }
    __syncthreads();

    // gather fragments per ISA 7.12.2 layouts and issue 4 WMMAs
    v16h af[2], bf[2];
#pragma unroll
    for (int i = 0; i < 2; ++i) {
      const int r = wm * 32 + i * 16 + rc;
      v8h lo = *(const v8h*)&sA[r][lh * 8];        // K: 8*lh + 0..7
      v8h hi = *(const v8h*)&sA[r][16 + lh * 8];   // K: 16 + 8*lh + 0..7
      af[i] = __builtin_shufflevector(lo, hi, 0,1,2,3,4,5,6,7,8,9,10,11,12,13,14,15);
    }
#pragma unroll
    for (int j = 0; j < 2; ++j) {
      const int n = wn * 32 + j * 16 + rc;
      v8h lo = *(const v8h*)&sB[n][lh * 8];        // K: 8*lh + 0..7
      v8h hi = *(const v8h*)&sB[n][16 + lh * 8];   // K: 16 + 8*lh + 0..7
      bf[j] = __builtin_shufflevector(lo, hi, 0,1,2,3,4,5,6,7,8,9,10,11,12,13,14,15);
    }
#pragma unroll
    for (int i = 0; i < 2; ++i)
#pragma unroll
      for (int j = 0; j < 2; ++j)
        acc[i][j] = __builtin_amdgcn_wmma_f32_16x16x32_f16(
            false, af[i], false, bf[j], (short)0, acc[i][j], false, false);
    __syncthreads();
  }

  if (!ADA) {
#pragma unroll
    for (int i = 0; i < 2; ++i)
#pragma unroll
      for (int j = 0; j < 2; ++j)
#pragma unroll
        for (int v = 0; v < 8; ++v) {
          const int row = bm0 + wm * 32 + i * 16 + lh * 8 + v;
          const int col = bn0 + wn * 32 + j * 16 + rc;
          float val = acc[i][j][v];
          if (bias) val += bias[col];
          const size_t idx = (size_t)row * ldc + col;
          if (accum) val += Cg[idx];
          Cg[idx] = val;
        }
  } else {
#pragma unroll
    for (int i = 0; i < 2; ++i)
#pragma unroll
      for (int v = 0; v < 8; ++v) {
        const int row = bm0 + wm * 32 + i * 16 + lh * 8 + v;
        const int b = row / S_;
        const int s = row % S_;
        float p = 0.0f;
#pragma unroll
        for (int j = 0; j < 2; ++j) {
          const int col = bn0 + wn * 32 + j * 16 + rc;
          float t = tanhf(acc[i][j][v] + (bias ? bias[col] : 0.0f) +
                          ho_embed[b * R_ + col]);
          p += t * ada_w[col];
        }
        // reduce over the 32 columns (2 per lane) held by this 16-lane group
        p += __shfl_xor(p, 1, 32);
        p += __shfl_xor(p, 2, 32);
        p += __shfl_xor(p, 4, 32);
        p += __shfl_xor(p, 8, 32);
        if (rc == 0) atomicAdd(&e_ada[b * (S_ + 1) + 1 + s], p);
      }
  }
}

// ---------------------------------------------------------------------------
// Elementwise / reduction kernels
// ---------------------------------------------------------------------------
__global__ void lstm_pointwise(const float* __restrict__ g,
                               const float* __restrict__ bih,
                               const float* __restrict__ bhh,
                               const float* __restrict__ cprev,
                               float* __restrict__ hout,
                               float* __restrict__ cout)
{
  int idx = blockIdx.x * blockDim.x + threadIdx.x;
  if (idx >= B_ * R_) return;
  int b = idx >> 10, r = idx & 1023;
  const float* gb = g + (size_t)b * 4 * R_;
  float i_ = gb[r]            + bih[r]            + bhh[r];
  float f_ = gb[R_ + r]       + bih[R_ + r]       + bhh[R_ + r];
  float gg = gb[2 * R_ + r]   + bih[2 * R_ + r]   + bhh[2 * R_ + r];
  float o_ = gb[3 * R_ + r]   + bih[3 * R_ + r]   + bhh[3 * R_ + r];
  float c2 = sigm(f_) * cprev[idx] + sigm(i_) * tanhf(gg);
  cout[idx] = c2;
  hout[idx] = sigm(o_) * tanhf(c2);
}

__global__ void st_kernel(const float* __restrict__ n5,
                          const float* __restrict__ b1,
                          const float* __restrict__ b2,
                          const float* __restrict__ c_lang,
                          float* __restrict__ st)
{
  int idx = blockIdx.x * blockDim.x + threadIdx.x;
  if (idx >= B_ * R_) return;
  int r = idx & 1023;
  st[idx] = sigm(n5[idx] + b1[r] + b2[r]) * tanhf(c_lang[idx]);
}

__global__ void relu_kernel(const float* __restrict__ x, float* __restrict__ y, int n) {
  int i = blockIdx.x * blockDim.x + threadIdx.x;
  if (i < n) y[i] = fmaxf(x[i], 0.0f);
}

__global__ void tanh_kernel(const float* __restrict__ x, float* __restrict__ y, int n) {
  int i = blockIdx.x * blockDim.x + threadIdx.x;
  if (i < n) y[i] = tanhf(x[i]);
}

__global__ void zero_kernel(float* __restrict__ x, int n) {
  int i = blockIdx.x * blockDim.x + threadIdx.x;
  if (i < n) x[i] = 0.0f;
}

// e[b,s] = sum_a tanh(p[b,s,a] + h[b,a]) * aw[a] + ab   (one wave per row)
__global__ void attention_e(const float* __restrict__ p_att,
                            const float* __restrict__ attH,
                            const float* __restrict__ aw,
                            const float* __restrict__ ab,
                            float* __restrict__ e, int S)
{
  int row = (blockIdx.x * blockDim.x + threadIdx.x) >> 5;
  int lane = threadIdx.x & 31;
  if (row >= B_ * S) return;
  int b = row / S;
  const float* p = p_att + (size_t)row * A_;
  const float* h = attH + (size_t)b * A_;
  float s = 0.0f;
  for (int a = lane; a < A_; a += 32) s += tanhf(p[a] + h[a]) * aw[a];
  s += __shfl_xor(s, 1, 32);
  s += __shfl_xor(s, 2, 32);
  s += __shfl_xor(s, 4, 32);
  s += __shfl_xor(s, 8, 32);
  s += __shfl_xor(s, 16, 32);
  if (lane == 0) e[row] = s + ab[0];
}

__device__ __forceinline__ float blockMax(float v, float* red) {
  int t = threadIdx.x;
  red[t] = v; __syncthreads();
  for (int s = 128; s > 0; s >>= 1) {
    if (t < s) red[t] = fmaxf(red[t], red[t + s]);
    __syncthreads();
  }
  float r = red[0]; __syncthreads();
  return r;
}
__device__ __forceinline__ float blockSum(float v, float* red) {
  int t = threadIdx.x;
  red[t] = v; __syncthreads();
  for (int s = 128; s > 0; s >>= 1) {
    if (t < s) red[t] += red[t + s];
    __syncthreads();
  }
  float r = red[0]; __syncthreads();
  return r;
}

// per-b masked softmax over S + weighted sum over feats -> out (B,R)
__global__ __launch_bounds__(256) void attn_softmax_wsum(
    const float* __restrict__ e, const float* __restrict__ mask,
    const float* __restrict__ feats, float* __restrict__ out, int S)
{
  __shared__ float w[256];
  __shared__ float red[256];
  int b = blockIdx.x, tid = threadIdx.x;
  float m = -3.4e38f;
  for (int s = tid; s < S; s += 256) m = fmaxf(m, e[b * S + s]);
  m = blockMax(m, red);
  float sum = 0.0f;
  for (int s = tid; s < S; s += 256) {
    float p = expf(e[b * S + s] - m) * mask[b * S + s];
    w[s] = p;
    sum += p;
  }
  sum = blockSum(sum, red);
  float inv = 1.0f / sum;
  for (int d = tid; d < R_; d += 256) {
    const float* f = feats + (size_t)b * S * R_ + d;
    float a = 0.0f;
    for (int s = 0; s < S; ++s) a += w[s] * f[(size_t)s * R_];
    out[b * R_ + d] = a * inv;
  }
}

// e_ada[b*(S+1)+0] = sum_r tanh(fr_embed + ho_embed)*ada_w   (one wave per b)
__global__ void ada_slot0(const float* __restrict__ fr_embed,
                          const float* __restrict__ ho_embed,
                          const float* __restrict__ ada_w,
                          float* __restrict__ e_ada)
{
  int b = (blockIdx.x * blockDim.x + threadIdx.x) >> 5;
  int lane = threadIdx.x & 31;
  if (b >= B_) return;
  float s = 0.0f;
  for (int r = lane; r < R_; r += 32)
    s += tanhf(fr_embed[b * R_ + r] + ho_embed[b * R_ + r]) * ada_w[r];
  s += __shfl_xor(s, 1, 32);
  s += __shfl_xor(s, 2, 32);
  s += __shfl_xor(s, 4, 32);
  s += __shfl_xor(s, 8, 32);
  s += __shfl_xor(s, 16, 32);
  if (lane == 0) e_ada[b * (S_ + 1)] = s;
}

// adaptive softmax over S+1, visAtt over [fake_region; att_feats], c_t_hat
__global__ __launch_bounds__(256) void ada_softmax_ct(
    const float* __restrict__ e, const float* __restrict__ fake_region,
    const float* __restrict__ att_feats, float* __restrict__ ct)
{
  __shared__ float pi[256];
  __shared__ float red[256];
  const int N = S_ + 1;  // 197
  int b = blockIdx.x, tid = threadIdx.x;
  float m = -3.4e38f;
  for (int s = tid; s < N; s += 256) m = fmaxf(m, e[b * N + s]);
  m = blockMax(m, red);
  float sum = 0.0f;
  for (int s = tid; s < N; s += 256) {
    float p = expf(e[b * N + s] - m);
    pi[s] = p;
    sum += p;
  }
  sum = blockSum(sum, red);
  float inv = 1.0f / sum;
  for (int s = tid; s < N; s += 256) pi[s] *= inv;
  __syncthreads();
  float beta = pi[S_];  // PI[:, -1]
  for (int d = tid; d < R_; d += 256) {
    float fr = fake_region[b * R_ + d];
    float acc = pi[0] * fr;
    const float* f = att_feats + (size_t)b * S_ * R_ + d;
    for (int s = 0; s < S_; ++s) acc += pi[s + 1] * f[(size_t)s * R_];
    ct[b * R_ + d] = beta * fr + (1.0f - beta) * acc;
  }
}

// ---------------------------------------------------------------------------
// Host
// ---------------------------------------------------------------------------
static void GEMM(hipStream_t stream, const float* A, int lda, const float* W, int ldw,
                 float* C, int ldc, const float* bias, int accum, int M, int N, int K)
{
  dim3 grid(N / 128, M / 64);
  gemm_kernel<false><<<grid, dim3(256), 0, stream>>>(
      A, lda, W, ldw, C, ldc, bias, accum, M, N, K, nullptr, nullptr, nullptr);
}

extern "C" void kernel_launch(void* const* d_in, const int* in_sizes, int n_in,
                              void* d_out, int out_size, void* d_ws, size_t ws_size,
                              hipStream_t stream)
{
  (void)in_sizes; (void)n_in; (void)out_size; (void)ws_size;
  const float* xt          = (const float*)d_in[0];
  const float* state_h     = (const float*)d_in[1];
  const float* state_c     = (const float*)d_in[2];
  const float* fc_feats    = (const float*)d_in[3];
  const float* att_feats   = (const float*)d_in[4];
  const float* obj_feats   = (const float*)d_in[5];
  const float* attr_feats  = (const float*)d_in[6];
  const float* rela_feats  = (const float*)d_in[7];
  const float* p_obj       = (const float*)d_in[8];
  const float* p_attr      = (const float*)d_in[9];
  const float* p_rela      = (const float*)d_in[10];
  const float* att_masks   = (const float*)d_in[11];
  const float* rela_masks  = (const float*)d_in[12];
  const float* att_wih     = (const float*)d_in[13];
  const float* att_whh     = (const float*)d_in[14];
  const float* att_bih     = (const float*)d_in[15];
  const float* att_bhh     = (const float*)d_in[16];
  const float* lang_wih    = (const float*)d_in[17];
  const float* lang_whh    = (const float*)d_in[18];
  const float* lang_bih    = (const float*)d_in[19];
  const float* lang_bhh    = (const float*)d_in[20];
  const float* xt2_w       = (const float*)d_in[21];
  const float* xt2_b       = (const float*)d_in[22];
  const float* h_att2_w    = (const float*)d_in[23];
  const float* h_att2_b    = (const float*)d_in[24];
  const float* r_i2h_w     = (const float*)d_in[25];
  const float* r_i2h_b     = (const float*)d_in[26];
  const float* r_h2h_w     = (const float*)d_in[27];
  const float* r_h2h_b     = (const float*)d_in[28];
  const float* huu_w       = (const float*)d_in[29];
  const float* huu_b       = (const float*)d_in[30];
  const float* obj_h2att_w = (const float*)d_in[31];
  const float* obj_h2att_b = (const float*)d_in[32];
  const float* obj_alpha_w = (const float*)d_in[33];
  const float* obj_alpha_b = (const float*)d_in[34];
  const float* attr_h2att_w = (const float*)d_in[35];
  const float* attr_h2att_b = (const float*)d_in[36];
  const float* attr_alpha_w = (const float*)d_in[37];
  const float* attr_alpha_b = (const float*)d_in[38];
  const float* rela_h2att_w = (const float*)d_in[39];
  const float* rela_h2att_b = (const float*)d_in[40];
  const float* rela_alpha_w = (const float*)d_in[41];
  const float* rela_alpha_b = (const float*)d_in[42];
  const float* fr_lin_w    = (const float*)d_in[43];
  const float* fr_lin_b    = (const float*)d_in[44];
  const float* fr_emb_w    = (const float*)d_in[45];
  const float* fr_emb_b    = (const float*)d_in[46];
  const float* ho_lin_w    = (const float*)d_in[47];
  const float* ho_lin_b    = (const float*)d_in[48];
  const float* ho_emb_w    = (const float*)d_in[49];
  const float* ho_emb_b    = (const float*)d_in[50];
  const float* ada_w       = (const float*)d_in[51];
  const float* ada_b       = (const float*)d_in[52];  (void)ada_b; // constant shift, cancels in softmax
  const float* cfe_w       = (const float*)d_in[53];
  const float* cfe_b       = (const float*)d_in[54];

  const int BR = B_ * R_;
  float* out    = (float*)d_out;
  float* out_y  = out;            // output (B,R)
  float* h_att  = out + 1 * BR;   // new_h[0]
  float* h_lang = out + 2 * BR;   // new_h[1]
  float* c_att  = out + 3 * BR;   // new_c[0]
  float* c_lang = out + 4 * BR;   // new_c[1]

  // workspace carve (fp32)
  float* wsf = (float*)d_ws;
  size_t off = 0;
  auto alloc = [&](size_t n) { float* p = wsf + off; off += (n + 63) & ~(size_t)63; return p; };
  float* g_att    = alloc((size_t)B_ * 4 * R_);
  float* g_lang   = alloc((size_t)B_ * 4 * R_);
  float* h_att_raw= alloc(BR);
  float* attH     = alloc((size_t)B_ * A_);
  float* e_obj    = alloc((size_t)B_ * S_);
  float* e_attr   = alloc((size_t)B_ * S_);
  float* e_rela   = alloc((size_t)B_ * SR_);
  float* att_obj  = alloc(BR);
  float* att_attr = alloc(BR);
  float* att_rela = alloc(BR);
  float* n5       = alloc(BR);
  float* st_buf   = alloc(BR);
  float* fr_raw   = alloc(BR);
  float* fake_reg = alloc(BR);
  float* fr_embed = alloc(BR);
  float* ho_raw   = alloc(BR);
  float* ho_buf   = alloc(BR);
  float* ho_embed = alloc(BR);
  float* e_ada    = alloc((size_t)B_ * (S_ + 1));
  float* ct       = alloc(BR);

  const int EW = (BR + 255) / 256;

  // ---- attention LSTM: g = [prev_h, fc, xt] @ wih^T + h0 @ whh^T (K-sliced) ----
  GEMM(stream, state_h + BR, R_, att_wih + 0,      3 * R_, g_att, 4 * R_, nullptr, 0, B_, 4 * R_, R_);
  GEMM(stream, fc_feats,     R_, att_wih + R_,     3 * R_, g_att, 4 * R_, nullptr, 1, B_, 4 * R_, R_);
  GEMM(stream, xt,           R_, att_wih + 2 * R_, 3 * R_, g_att, 4 * R_, nullptr, 1, B_, 4 * R_, R_);
  GEMM(stream, state_h,      R_, att_whh,          R_,     g_att, 4 * R_, nullptr, 1, B_, 4 * R_, R_);
  lstm_pointwise<<<EW, 256, 0, stream>>>(g_att, att_bih, att_bhh, state_c, h_att_raw, c_att);

  // ---- h_att = xt@xt2 + b + h_att_raw@h_att2 + b  (into d_out new_h[0]) ----
  GEMM(stream, xt,        R_, xt2_w,    R_, h_att, R_, xt2_b,    0, B_, R_, R_);
  GEMM(stream, h_att_raw, R_, h_att2_w, R_, h_att, R_, h_att2_b, 1, B_, R_, R_);

  // ---- three attentions ----
  GEMM(stream, h_att, R_, obj_h2att_w, R_, attH, A_, obj_h2att_b, 0, B_, A_, R_);
  attention_e<<<(B_ * S_ + 7) / 8, 256, 0, stream>>>(p_obj, attH, obj_alpha_w, obj_alpha_b, e_obj, S_);
  attn_softmax_wsum<<<B_, 256, 0, stream>>>(e_obj, att_masks, obj_feats, att_obj, S_);

  GEMM(stream, h_att, R_, attr_h2att_w, R_, attH, A_, attr_h2att_b, 0, B_, A_, R_);
  attention_e<<<(B_ * S_ + 7) / 8, 256, 0, stream>>>(p_attr, attH, attr_alpha_w, attr_alpha_b, e_attr, S_);
  attn_softmax_wsum<<<B_, 256, 0, stream>>>(e_attr, att_masks, attr_feats, att_attr, S_);

  GEMM(stream, h_att, R_, rela_h2att_w, R_, attH, A_, rela_h2att_b, 0, B_, A_, R_);
  attention_e<<<(B_ * SR_ + 7) / 8, 256, 0, stream>>>(p_rela, attH, rela_alpha_w, rela_alpha_b, e_rela, SR_);
  attn_softmax_wsum<<<B_, 256, 0, stream>>>(e_rela, rela_masks, rela_feats, att_rela, SR_);

  // ---- language LSTM: lang_in = [h_att, att_obj, att_attr, att_rela] ----
  GEMM(stream, h_att,        R_, lang_wih + 0,      4 * R_, g_lang, 4 * R_, nullptr, 0, B_, 4 * R_, R_);
  GEMM(stream, att_obj,      R_, lang_wih + R_,     4 * R_, g_lang, 4 * R_, nullptr, 1, B_, 4 * R_, R_);
  GEMM(stream, att_attr,     R_, lang_wih + 2 * R_, 4 * R_, g_lang, 4 * R_, nullptr, 1, B_, 4 * R_, R_);
  GEMM(stream, att_rela,     R_, lang_wih + 3 * R_, 4 * R_, g_lang, 4 * R_, nullptr, 1, B_, 4 * R_, R_);
  GEMM(stream, state_h + BR, R_, lang_whh,          R_,     g_lang, 4 * R_, nullptr, 1, B_, 4 * R_, R_);
  lstm_pointwise<<<EW, 256, 0, stream>>>(g_lang, lang_bih, lang_bhh, state_c + BR, h_lang, c_lang);

  // ---- n5 / st ----
  GEMM(stream, h_att,        R_, r_i2h_w + 0,      4 * R_, n5, R_, nullptr, 0, B_, R_, R_);
  GEMM(stream, att_obj,      R_, r_i2h_w + R_,     4 * R_, n5, R_, nullptr, 1, B_, R_, R_);
  GEMM(stream, att_attr,     R_, r_i2h_w + 2 * R_, 4 * R_, n5, R_, nullptr, 1, B_, R_, R_);
  GEMM(stream, att_rela,     R_, r_i2h_w + 3 * R_, 4 * R_, n5, R_, nullptr, 1, B_, R_, R_);
  GEMM(stream, state_h + BR, R_, r_h2h_w,          R_,     n5, R_, nullptr, 1, B_, R_, R_);
  st_kernel<<<EW, 256, 0, stream>>>(n5, r_i2h_b, r_h2h_b, c_lang, st_buf);

  // ---- fake region / embeddings ----
  GEMM(stream, st_buf, R_, fr_lin_w, R_, fr_raw, R_, fr_lin_b, 0, B_, R_, R_);
  relu_kernel<<<EW, 256, 0, stream>>>(fr_raw, fake_reg, BR);
  GEMM(stream, fake_reg, R_, fr_emb_w, R_, fr_embed, R_, fr_emb_b, 0, B_, R_, R_);
  GEMM(stream, h_lang, R_, ho_lin_w, R_, ho_raw, R_, ho_lin_b, 0, B_, R_, R_);
  tanh_kernel<<<EW, 256, 0, stream>>>(ho_raw, ho_buf, BR);
  GEMM(stream, ho_buf, R_, ho_emb_w, R_, ho_embed, R_, ho_emb_b, 0, B_, R_, R_);

  // ---- adaptive attention logits (fused cfe GEMM, cfe never materialized) ----
  const int nEada = B_ * (S_ + 1);
  zero_kernel<<<(nEada + 255) / 256, 256, 0, stream>>>(e_ada, nEada);
  ada_slot0<<<(B_ * 32 + 255) / 256, 256, 0, stream>>>(fr_embed, ho_embed, ada_w, e_ada);
  {
    dim3 grid(R_ / 128, (B_ * S_) / 64);
    gemm_kernel<true><<<grid, dim3(256), 0, stream>>>(
        p_obj, A_, cfe_w, A_, nullptr, 0, cfe_b, 0, B_ * S_, R_, A_,
        ho_embed, ada_w, e_ada);
  }
  ada_softmax_ct<<<B_, 256, 0, stream>>>(e_ada, fake_reg, att_feats, ct);

  // ---- output = [h_att, h_lang, c_t_hat] @ huu^T + b ----
  GEMM(stream, h_att,  R_, huu_w + 0,      3 * R_, out_y, R_, huu_b,  0, B_, R_, R_);
  GEMM(stream, h_lang, R_, huu_w + R_,     3 * R_, out_y, R_, nullptr, 1, B_, R_, R_);
  GEMM(stream, ct,     R_, huu_w + 2 * R_, 3 * R_, out_y, R_, nullptr, 1, B_, R_, R_);
}
